// DGLJTMPN_7988639171235
// MI455X (gfx1250) — matrix operand order repaired and checked by reading, not compiled, not run on hardware
//
#include <hip/hip_runtime.h>

typedef __attribute__((ext_vector_type(16))) __bf16 v16bf;
typedef __attribute__((ext_vector_type(8)))  __bf16 v8bf;
typedef __attribute__((ext_vector_type(8)))  float  v8f;
typedef __attribute__((ext_vector_type(4)))  int    v4i;

#define HID 256

// ---------------------------------------------------------------------------
// Cooperative stage of the (pre-swizzled) B matrix into LDS.
// Uses CDNA5 async global->LDS copies when the toolchain exposes them,
// otherwise falls back to global_load_b128 + ds_store_b128.
// All threads of the block participate; ends with a workgroup barrier.
// ---------------------------------------------------------------------------
__device__ __forceinline__ void stage_b_to_lds(const __bf16* __restrict__ src,
                                               __bf16* lds, int nbytes)
{
    const int tid    = threadIdx.x;
    const int nthr   = blockDim.x;
    const int chunks = nbytes >> 4;                    // 16-byte chunks
#if __has_builtin(__builtin_amdgcn_global_load_async_to_lds_b128)
    typedef __attribute__((address_space(1))) v4i* as1_v4i_ptr;
    typedef __attribute__((address_space(3))) v4i* as3_v4i_ptr;
    as1_v4i_ptr g = (as1_v4i_ptr)src;
    as3_v4i_ptr l = (as3_v4i_ptr)lds;
    for (int i = tid; i < chunks; i += nthr)
        __builtin_amdgcn_global_load_async_to_lds_b128(g + i, l + i, 0, 0);
#if __has_builtin(__builtin_amdgcn_s_wait_asynccnt)
    __builtin_amdgcn_s_wait_asynccnt(0);
#else
    asm volatile("s_wait_asynccnt 0x0" ::: "memory");
#endif
#else
    for (int i = tid; i < chunks; i += nthr)
        ((uint4*)lds)[i] = ((const uint4*)src)[i];
#endif
    __syncthreads();
}

// Read one B fragment (lane-private 32B) from LDS as two 16B loads.
__device__ __forceinline__ v16bf load_b_frag(const __bf16* lds, int kt, int nt, int lane)
{
    const __bf16* bp = lds + ((size_t)(kt * 16 + nt) * 32 + lane) * 16;
    v8bf b0 = *(const v8bf*)(bp);
    v8bf b1 = *(const v8bf*)(bp + 8);
    v16bf b;
#pragma unroll
    for (int j = 0; j < 8; ++j) { b[j] = b0[j]; b[j + 8] = b1[j]; }
    return b;
}

// ---------------------------------------------------------------------------
// Swizzle a row-major fp32 weight [Krows, 256] into the WMMA bf16 B-operand
// layout: Bs[((kt*16+nt)*32 + lane)*16 + j] = W[kt*32 + (lane>>4)*8 + (j<8 ? j : j+8)][nt*16 + (lane&15)]
// ---------------------------------------------------------------------------
__global__ void swizzleB_kernel(const float* __restrict__ W, __bf16* __restrict__ Bs,
                                int Krows)
{
    const int kt   = blockIdx.x >> 4;
    const int nt   = blockIdx.x & 15;
    const int lane = threadIdx.x;
    const int n    = nt * 16 + (lane & 15);
    const int kb   = kt * 32 + (lane >> 4) * 8;
    __bf16* dst = Bs + ((size_t)(kt * 16 + nt) * 32 + lane) * 16;
#pragma unroll
    for (int j = 0; j < 16; ++j) {
        const int k = kb + (j < 8 ? j : j + 8);
        const float v = (k < Krows) ? W[(size_t)k * HID + n] : 0.0f;
        dst[j] = (__bf16)v;
    }
}

// ---------------------------------------------------------------------------
// Main loop GEMM: msg = relu(msg_input + (accum + node_alpha[edge_src]) @ W_h)
// 8 waves/block; W_h (128KB bf16, swizzled) staged in LDS once per block;
// one wave per 16-row tile; 16 n-tiles x 8 k-steps of v_wmma_f32_16x16x32_bf16.
// ---------------------------------------------------------------------------
__global__ void __launch_bounds__(256)
gemm_wh_kernel(const float* __restrict__ accum, const float* __restrict__ node_alpha,
               const int* __restrict__ edge_src, const float* __restrict__ msg_input,
               const __bf16* __restrict__ Whs, float* __restrict__ msg_out, int M)
{
    extern __shared__ __bf16 ldsB[];                  // 256*256 bf16 = 128KB
    stage_b_to_lds(Whs, ldsB, 256 * HID * 2);

    const int wave = (int)(threadIdx.x >> 5);
    const int lane = (int)(threadIdx.x & 31);
    const int tile = blockIdx.x * 8 + wave;
    if (tile >= (M >> 4)) return;                     // wave-uniform (after barrier)
    const int l16  = lane & 15;
    const int half = lane >> 4;
    const int r    = tile * 16 + l16;
    const int src  = edge_src[r];
    const float* xa = accum      + (size_t)r   * HID;
    const float* xb = node_alpha + (size_t)src * HID;

    // A fragments: lane holds row (lane&15), K = {h*8..h*8+7} U {16+h*8..} per k-step
    v16bf Af[8];
#pragma unroll
    for (int kt = 0; kt < 8; ++kt) {
        const int k0 = kt * 32 + half * 8;
        float va[16], vb[16];
        ((float4*)va)[0] = *(const float4*)(xa + k0);
        ((float4*)va)[1] = *(const float4*)(xa + k0 + 4);
        ((float4*)va)[2] = *(const float4*)(xa + k0 + 16);
        ((float4*)va)[3] = *(const float4*)(xa + k0 + 20);
        ((float4*)vb)[0] = *(const float4*)(xb + k0);
        ((float4*)vb)[1] = *(const float4*)(xb + k0 + 4);
        ((float4*)vb)[2] = *(const float4*)(xb + k0 + 16);
        ((float4*)vb)[3] = *(const float4*)(xb + k0 + 20);
#pragma unroll
        for (int j = 0; j < 16; ++j) Af[kt][j] = (__bf16)(va[j] + vb[j]);
    }

    const size_t orow = (size_t)(tile * 16 + half * 8) * HID + l16;
    for (int nt = 0; nt < 16; ++nt) {
        v8f c;
        const float* cp = msg_input + orow + nt * 16;
#pragma unroll
        for (int i = 0; i < 8; ++i) c[i] = cp[(size_t)i * HID];
#pragma unroll
        for (int kt = 0; kt < 8; ++kt) {
            v16bf b = load_b_frag(ldsB, kt, nt, lane);
            c = __builtin_amdgcn_wmma_f32_16x16x32_bf16(false, Af[kt], false, b,
                                                        (short)0, c, false, false);
        }
        float* op = msg_out + orow + nt * 16;
#pragma unroll
        for (int i = 0; i < 8; ++i) op[(size_t)i * HID] = fmaxf(c[i], 0.0f);
    }
}

// ---------------------------------------------------------------------------
// Generic bf16 GEMM: out[M,256] = A[M, KSTEPS*32](bf16) @ Bs (+bias) (relu)
// B staged in LDS (KSTEPS*32*256 bf16). SCATTER mode: atomicAdd into
// out[gids[row]*256 + n] (fused graph-mean sums). out2: optional relu copy.
// ---------------------------------------------------------------------------
template<int KSTEPS, bool RELU_OUT, bool HAS_BIAS, bool SCATTER>
__global__ void __launch_bounds__(256)
gemm_bf16_kernel(const __bf16* __restrict__ A, const __bf16* __restrict__ Bs,
                 const float* __restrict__ bias, float* __restrict__ out,
                 const int* __restrict__ gids, float* __restrict__ out2, int M)
{
    extern __shared__ __bf16 ldsB[];
    stage_b_to_lds(Bs, ldsB, KSTEPS * 32 * HID * 2);

    const int wave = (int)(threadIdx.x >> 5);
    const int lane = (int)(threadIdx.x & 31);
    const int tile = blockIdx.x * 8 + wave;
    if (tile >= (M >> 4)) return;
    const int l16  = lane & 15;
    const int half = lane >> 4;
    const int r    = tile * 16 + l16;

    v16bf Af[KSTEPS];
    const __bf16* ar = A + (size_t)r * (KSTEPS * 32);
#pragma unroll
    for (int kt = 0; kt < KSTEPS; ++kt) {
        const int k0 = kt * 32 + half * 8;
        v8bf s0 = *(const v8bf*)(ar + k0);
        v8bf s1 = *(const v8bf*)(ar + k0 + 16);
#pragma unroll
        for (int j = 0; j < 8; ++j) { Af[kt][j] = s0[j]; Af[kt][j + 8] = s1[j]; }
    }

    int gid[8];
    if constexpr (SCATTER) {
#pragma unroll
        for (int i = 0; i < 8; ++i) gid[i] = gids[tile * 16 + half * 8 + i];
    }
    const size_t orow = (size_t)(tile * 16 + half * 8) * HID + l16;

    for (int nt = 0; nt < 16; ++nt) {
        const float binit = HAS_BIAS ? bias[nt * 16 + l16] : 0.0f;
        v8f c;
#pragma unroll
        for (int i = 0; i < 8; ++i) c[i] = binit;
#pragma unroll
        for (int kt = 0; kt < KSTEPS; ++kt) {
            v16bf b = load_b_frag(ldsB, kt, nt, lane);
            c = __builtin_amdgcn_wmma_f32_16x16x32_bf16(false, Af[kt], false, b,
                                                        (short)0, c, false, false);
        }
        if constexpr (SCATTER) {
#pragma unroll
            for (int i = 0; i < 8; ++i) {
                float v = c[i];
                if (RELU_OUT) v = fmaxf(v, 0.0f);
                atomicAdd(out + (size_t)gid[i] * HID + nt * 16 + l16, v);
            }
        } else {
            float* op = out + orow + nt * 16;
#pragma unroll
            for (int i = 0; i < 8; ++i) {
                const float v = c[i];
                op[(size_t)i * HID] = RELU_OUT ? fmaxf(v, 0.0f) : v;
                if (out2) out2[orow + nt * 16 + (size_t)i * HID] = fmaxf(v, 0.0f);
            }
        }
    }
}

// ---------------------------------------------------------------------------
// Elementwise / scatter helpers
// ---------------------------------------------------------------------------
__global__ void pack_feat_kernel(const float* __restrict__ node_x, const float* __restrict__ bond_x,
                                 const int* __restrict__ esrc, __bf16* __restrict__ Afeat, int E)
{
    const int e = blockIdx.x;          // one edge per block
    const int c = threadIdx.x;         // 64 cols (K padded 40 -> 64)
    if (e >= E) return;
    float v = 0.0f;
    if (c < 35)      v = node_x[(size_t)esrc[e] * 35 + c];
    else if (c < 40) v = bond_x[(size_t)e * 5 + (c - 35)];
    Afeat[(size_t)e * 64 + c] = (__bf16)v;
}

__global__ void pack_fin_kernel(const float* __restrict__ node_x, const float* __restrict__ m,
                                const float* __restrict__ node_alpha, __bf16* __restrict__ Afin,
                                int total)
{
    const int idx = blockIdx.x * blockDim.x + threadIdx.x;
    if (idx >= total) return;
    const int col = idx % 320;         // K padded 291 -> 320
    const int row = idx / 320;
    float v = 0.0f;
    if (col < 35)       v = node_x[(size_t)row * 35 + col];
    else if (col < 291) {
        const int c = col - 35;
        v = m[(size_t)row * HID + c] + node_alpha[(size_t)row * HID + c];
    }
    Afin[idx] = (__bf16)v;
}

__global__ void scatter_rows_kernel(const float* __restrict__ srcmat, const int* __restrict__ idx,
                                    float* __restrict__ dst, int S)
{
    const int e = blockIdx.x;
    const int c = threadIdx.x;         // 256
    if (e >= S) return;
    atomicAdd(dst + (size_t)idx[e] * HID + c, srcmat[(size_t)e * HID + c]);
}

__global__ void scatter_lg_kernel(const float* __restrict__ msg, const int* __restrict__ lsrc,
                                  const int* __restrict__ ldst, float* __restrict__ accum, int S)
{
    const int e = blockIdx.x;
    const int c = threadIdx.x;         // 256
    if (e >= S) return;
    atomicAdd(accum + (size_t)ldst[e] * HID + c, msg[(size_t)lsrc[e] * HID + c]);
}

__global__ void count_kernel(const int* __restrict__ gids, float* __restrict__ counts, int N)
{
    const int i = blockIdx.x * blockDim.x + threadIdx.x;
    if (i < N) atomicAdd(counts + gids[i], 1.0f);
}

__global__ void finalize_kernel(const float* __restrict__ gsum, const float* __restrict__ counts,
                                float* __restrict__ out, int total)
{
    const int i = blockIdx.x * blockDim.x + threadIdx.x;
    if (i < total) out[i] = gsum[i] / fmaxf(counts[i >> 8], 1.0f);
}

// ---------------------------------------------------------------------------
extern "C" void kernel_launch(void* const* d_in, const int* in_sizes, int n_in,
                              void* d_out, int out_size, void* d_ws, size_t ws_size,
                              hipStream_t stream)
{
    const float* node_x    = (const float*)d_in[0];
    const float* bond_x    = (const float*)d_in[1];
    const float* tree_mess = (const float*)d_in[2];
    const float* W_i       = (const float*)d_in[3];
    const float* W_h       = (const float*)d_in[4];
    const float* W_o       = (const float*)d_in[5];
    const float* b_o       = (const float*)d_in[6];
    const int*   edge_src  = (const int*)d_in[7];
    const int*   edge_dst  = (const int*)d_in[8];
    const int*   lg_src    = (const int*)d_in[9];
    const int*   lg_dst    = (const int*)d_in[10];
    const int*   tgt_nodes = (const int*)d_in[11];
    const int*   graph_ids = (const int*)d_in[12];

    const int E  = in_sizes[7];    // 300000 (multiple of 16)
    const int N  = in_sizes[12];   // 150000 (multiple of 16)
    const int LG = in_sizes[9];    // 600000
    const int T  = in_sizes[11];   // 60000
    const int G  = out_size / HID; // 2048

    // --- workspace layout (time-aliased regions) ---
    char* ws = (char*)d_ws;
    size_t off = 0;
    auto alloc = [&](size_t bytes) -> char* {
        char* p = ws + off;
        off += (bytes + 255) & ~(size_t)255;
        return p;
    };
    float*  node_alpha = (float*)alloc((size_t)N * HID * 4);
    char*   regA       = alloc((size_t)E * HID * 4);   // msg_input, later Afin
    float*  msg_input  = (float*)regA;
    __bf16* Afin       = (__bf16*)regA;                // [N,320] bf16 (fits)
    float*  msg        = (float*)alloc((size_t)E * HID * 4);
    char*   regB       = alloc((size_t)E * HID * 4);   // Afeat first, then accum
    float*  accum      = (float*)regB;
    __bf16* Afeat      = (__bf16*)regB;                // [E,64] bf16 (fits)
    float*  mbuf       = (float*)alloc((size_t)N * HID * 4);
    __bf16* Wis        = (__bf16*)alloc((size_t)64  * HID * 2);
    __bf16* Whs        = (__bf16*)alloc((size_t)256 * HID * 2);
    __bf16* Wos        = (__bf16*)alloc((size_t)320 * HID * 2);
    float*  gsum       = (float*)alloc((size_t)G * HID * 4);
    float*  counts     = (float*)alloc((size_t)G * 4);
    (void)ws_size; (void)n_in;

    const int etiles = E >> 4, ntiles = N >> 4;
    const dim3 gblk(256);                               // 8 waves/block
    const dim3 ggridE((etiles + 7) / 8);
    const dim3 ggridN((ntiles + 7) / 8);

    // 1) Pre-swizzle weights into WMMA B layout (bf16).
    swizzleB_kernel<<<dim3((64  / 32) * 16), dim3(32), 0, stream>>>(W_i, Wis, 40);
    swizzleB_kernel<<<dim3((256 / 32) * 16), dim3(32), 0, stream>>>(W_h, Whs, 256);
    swizzleB_kernel<<<dim3((320 / 32) * 16), dim3(32), 0, stream>>>(W_o, Wos, 291);

    // 2) node_alpha = scatter_add(tree_mess at tgt_nodes)
    (void)hipMemsetAsync(node_alpha, 0, (size_t)N * HID * 4, stream);
    scatter_rows_kernel<<<dim3(T), dim3(HID), 0, stream>>>(tree_mess, tgt_nodes, node_alpha, T);

    // 3) msg_input = [node_x[src];bond_x] @ W_i ; msg = relu(msg_input)
    pack_feat_kernel<<<dim3(E), dim3(64), 0, stream>>>(node_x, bond_x, edge_src, Afeat, E);
    gemm_bf16_kernel<2, false, false, false><<<ggridE, gblk, 64 * HID * 2, stream>>>(
        Afeat, Wis, nullptr, msg_input, nullptr, msg, E);

    // 4) loopy BP: 3 iterations on the line graph
    for (int it = 0; it < 3; ++it) {
        (void)hipMemsetAsync(accum, 0, (size_t)E * HID * 4, stream);
        scatter_lg_kernel<<<dim3(LG), dim3(HID), 0, stream>>>(msg, lg_src, lg_dst, accum, LG);
        gemm_wh_kernel<<<ggridE, gblk, 256 * HID * 2, stream>>>(
            accum, node_alpha, edge_src, msg_input, Whs, msg, E);
    }

    // 5) m = segment_sum(msg over edge_dst)
    (void)hipMemsetAsync(mbuf, 0, (size_t)N * HID * 4, stream);
    scatter_rows_kernel<<<dim3(E), dim3(HID), 0, stream>>>(msg, edge_dst, mbuf, E);

    // 6) h = relu([node_x ; m+alpha] @ W_o + b_o), fused graph-sum scatter
    pack_fin_kernel<<<dim3((N * 320 + 255) / 256), dim3(256), 0, stream>>>(
        node_x, mbuf, node_alpha, Afin, N * 320);
    (void)hipMemsetAsync(gsum,   0, (size_t)G * HID * 4, stream);
    (void)hipMemsetAsync(counts, 0, (size_t)G * 4, stream);
    count_kernel<<<dim3((N + 255) / 256), dim3(256), 0, stream>>>(graph_ids, counts, N);
    gemm_bf16_kernel<10, true, true, true><<<ggridN, gblk, 320 * HID * 2, stream>>>(
        Afin, Wos, b_o, gsum, graph_ids, nullptr, N);

    // 7) mean per graph
    finalize_kernel<<<dim3((out_size + 255) / 256), dim3(256), 0, stream>>>(
        gsum, counts, (float*)d_out, out_size);
}
